// GNNRegressor_54503134986921
// MI455X (gfx1250) — compile-verified, matrix operand order checked by reading, not run on hardware
//
#include <hip/hip_runtime.h>
#include <hip/hip_bf16.h>
#include <math.h>

// ---------------------------------------------------------------------------
// GNN NNConv regressor for MI455X (gfx1250, wave32, WMMA).
// Dominant GEMM (edge-conditioned weights) is fused as a Khatri-Rao product
// GEMM: msg = (a1 (x) h[src]) @ W2'  with W2' [4160,64] preswizzled to bf16
// WMMA B-fragments (L2-resident). Theta is never materialized (saves ~2 GB
// of HBM traffic). A-fragments are built with v_pk_mul_bf16 (8 VALU / step)
// so the VALU stream co-executes cleanly with the XDL WMMA stream.
// ---------------------------------------------------------------------------

typedef __attribute__((ext_vector_type(16))) __bf16    v16bf;
typedef __attribute__((ext_vector_type(8)))  float     v8f;
typedef __attribute__((ext_vector_type(8)))  unsigned  v8u;

#define N_NODES  10000
#define N_EDGES  40000
#define N_GRAPHS 64
#define EDGE_DIM 16
#define LAYERS   3
// K-steps for fused message GEMM: 4096 (k,i pairs) + 64 (bias rows) = 4160 = 130*32
#define MSG_STEPS 130

// A-fragment K mapping for v_wmma_*_16x16x32 (16-bit A, wave32):
// lane<16: elems 0..7 -> K0..7, elems 8..15 -> K16..23
// lane>=16: elems 0..7 -> K8..15, elems 8..15 -> K24..31
__device__ __forceinline__ int kmap(int j, int hi) {
  return (j < 8 ? j : j + 8) + (hi ? 8 : 0);
}

__device__ __forceinline__ unsigned pack2_bf16(float lo, float hi) {
  union { __bf16 b[2]; unsigned u; } p;
  p.b[0] = (__bf16)lo;
  p.b[1] = (__bf16)hi;
  return p.u;
}

// ---------------------------- utility kernels ------------------------------

__global__ void k_zero(float* __restrict__ p, int n) {
  for (int i = blockIdx.x * blockDim.x + threadIdx.x; i < n; i += gridDim.x * blockDim.x)
    p[i] = 0.0f;
}

__global__ void k_deg(const int* __restrict__ dst, float* __restrict__ deg, int E) {
  int e = blockIdx.x * blockDim.x + threadIdx.x;
  if (e < E) atomicAdd(&deg[dst[e]], 1.0f);
}

__global__ void k_invdeg(const float* __restrict__ deg, float* __restrict__ inv, int N) {
  int n = blockIdx.x * blockDim.x + threadIdx.x;
  if (n < N) { float d = deg[n]; inv[n] = d > 0.0f ? 1.0f / d : 0.0f; }
}

// ------------------------- weight pre-swizzle ------------------------------
// Pack a [K,64] f32 matrix into per-lane bf16 WMMA B-fragments:
// layout Wpre[step][otile][lane][elem16] -> one contiguous 32B load per lane.

__global__ void k_prep_w64(const float* __restrict__ W, __bf16* __restrict__ Wpre) {
  int t = blockIdx.x * blockDim.x + threadIdx.x;           // 2*4*32*16 = 4096
  if (t >= 2 * 4 * 32 * 16) return;
  int j = t & 15, lane = (t >> 4) & 31, ot = (t >> 9) & 3, s = t >> 11;
  int kk = s * 32 + kmap(j, lane >> 4);
  int o  = ot * 16 + (lane & 15);
  Wpre[t] = (__bf16)W[kk * 64 + o];
}

// edge_w2 [64,4096] (+ edge_b2 [4096] as K rows 4096..4159) -> 130-step frags
__global__ void k_prep_w2(const float* __restrict__ w2, const float* __restrict__ b2,
                          __bf16* __restrict__ Bpre) {
  int t = blockIdx.x * blockDim.x + threadIdx.x;           // 130*4*32*16 = 266240
  if (t >= MSG_STEPS * 4 * 32 * 16) return;
  int j = t & 15, lane = (t >> 4) & 31, ot = (t >> 9) & 3, s = t >> 11;
  int kk = s * 32 + kmap(j, lane >> 4);
  int o  = ot * 16 + (lane & 15);
  float v;
  if (kk < 4096) { int k = kk >> 6, i = kk & 63; v = w2[k * 4096 + i * 64 + o]; }
  else           { int i = kk - 4096;            v = b2[i * 64 + o]; }
  Bpre[t] = (__bf16)v;
}

// --------------------------- edge MLP layer 1 ------------------------------
// a1 = silu(edge_attr @ W1 + b1), K=16 -> plain VALU, negligible FLOPs.
__global__ void k_edge_mlp1(const float* __restrict__ ea, const float* __restrict__ w1,
                            const float* __restrict__ b1, float* __restrict__ a1, int E) {
  int e = blockIdx.x;
  if (e >= E) return;
  int o = threadIdx.x;                                     // 64 threads
  float s = b1[o];
#pragma unroll
  for (int k = 0; k < EDGE_DIM; ++k) s += ea[e * EDGE_DIM + k] * w1[k * 64 + o];
  s = s / (1.0f + __expf(-s));
  a1[e * 64 + o] = s;
}

// ------------------------ fused message WMMA kernel ------------------------
// One wave per 16-edge tile. Within each 32-wide K chunk the a1 factor is a
// single scalar (chunks are 64-aligned sub-blocks), so the A-frag is
// broadcast(a1[e,k]) * cached-hs-window, built with 8x v_pk_mul_bf16.
// 130 steps x 4 WMMAs, then atomicAdd scatter into agg[dst].
__global__ __launch_bounds__(32)
void k_message(const float* __restrict__ a1, const float* __restrict__ h,
               const int* __restrict__ src, const int* __restrict__ dst,
               const __bf16* __restrict__ Bpre, float* __restrict__ agg, int E) {
  __shared__ float a1t[16 * 64];
  __shared__ float hst[16 * 64];
  __shared__ int   dstt[16];
  const int lane  = threadIdx.x;
  const int ebase = blockIdx.x * 16;

  for (int idx = lane; idx < 1024; idx += 32) {
    int r = idx >> 6, c = idx & 63;
    int e = ebase + r;
    a1t[idx] = (e < E) ? a1[(size_t)e * 64 + c] : 0.0f;
    hst[idx] = (e < E) ? h[(size_t)src[e] * 64 + c] : 0.0f;
  }
  if (lane < 16) dstt[lane] = (ebase + lane < E) ? dst[ebase + lane] : 0;
  __syncthreads();

  const int row = lane & 15;
  const int hi  = lane >> 4;

  // Cache this lane's hs contributions as packed bf16 pairs (A-frag element
  // pairs (2j,2j+1) map to consecutive i), for windows i in [0,32) / [32,64).
  unsigned h0p[8], h1p[8];
#pragma unroll
  for (int jp = 0; jp < 8; ++jp) {
    h0p[jp] = pack2_bf16(hst[row * 64 +      kmap(2 * jp, hi)],
                         hst[row * 64 +      kmap(2 * jp + 1, hi)]);
    h1p[jp] = pack2_bf16(hst[row * 64 + 32 + kmap(2 * jp, hi)],
                         hst[row * 64 + 32 + kmap(2 * jp + 1, hi)]);
  }

  v8f acc[4];
#pragma unroll
  for (int t = 0; t < 4; ++t) acc[t] = v8f{};

  const v16bf* bp = (const v16bf*)Bpre;
  for (int k = 0; k < 65; ++k) {            // 65 k-values x 2 windows = 130 steps
    // broadcast a1[e,k] as a packed bf16 pair; k==64 -> bias rows (A = hs * 1.0)
    unsigned akp;
    if (k < 64) { float ak = a1t[row * 64 + k]; akp = pack2_bf16(ak, ak); }
    else        { akp = 0x3F803F80u; }                    // {1.0bf16, 1.0bf16}
    {                                        // step 2k, window 0
      v8u afu;
#pragma unroll
      for (int jp = 0; jp < 8; ++jp) {
        unsigned r;
        asm("v_pk_mul_bf16 %0, %1, %2" : "=v"(r) : "v"(akp), "v"(h0p[jp]));
        afu[jp] = r;
      }
      v16bf af = __builtin_bit_cast(v16bf, afu);
      const v16bf* b = bp + ((2 * k) * 4) * 32 + lane;
#pragma unroll
      for (int t = 0; t < 4; ++t)
        acc[t] = __builtin_amdgcn_wmma_f32_16x16x32_bf16(false, af, false, b[t * 32],
                                                         (short)0, acc[t], false, false);
    }
    {                                        // step 2k+1, window 1
      v8u afu;
#pragma unroll
      for (int jp = 0; jp < 8; ++jp) {
        unsigned r;
        asm("v_pk_mul_bf16 %0, %1, %2" : "=v"(r) : "v"(akp), "v"(h1p[jp]));
        afu[jp] = r;
      }
      v16bf af = __builtin_bit_cast(v16bf, afu);
      const v16bf* b = bp + ((2 * k + 1) * 4) * 32 + lane;
#pragma unroll
      for (int t = 0; t < 4; ++t)
        acc[t] = __builtin_amdgcn_wmma_f32_16x16x32_bf16(false, af, false, b[t * 32],
                                                         (short)0, acc[t], false, false);
    }
  }

  // C/D layout: lane<16 -> rows 0..7 in vgpr 0..7, lane>=16 -> rows 8..15; col = lane&15
#pragma unroll
  for (int t = 0; t < 4; ++t) {
    int o = t * 16 + row;
#pragma unroll
    for (int v = 0; v < 8; ++v) {
      int m = v + hi * 8;
      int e = ebase + m;
      if (e < E) atomicAdd(&agg[(size_t)dstt[m] * 64 + o], acc[t][v]);
    }
  }
}

// -------------------- generic [N,64] @ [64,64] WMMA GEMM -------------------
// out = maybe_silu( in @ W + bias + add*scale ). 8 waves/block, 1 tile/wave.
__global__ __launch_bounds__(256)
void k_node_gemm(const float* __restrict__ in, const __bf16* __restrict__ Wpre,
                 const float* __restrict__ bias, const float* __restrict__ add,
                 const float* __restrict__ scale, float* __restrict__ out,
                 int N, int do_silu) {
  __shared__ float st[8][16 * 64];
  const int wave  = threadIdx.x >> 5;
  const int lane  = threadIdx.x & 31;
  const int nbase = (blockIdx.x * 8 + wave) * 16;
  float* my = st[wave];
  for (int idx = lane; idx < 1024; idx += 32) {
    int r = idx >> 6, c = idx & 63;
    int n = nbase + r;
    my[idx] = (n < N) ? in[(size_t)n * 64 + c] : 0.0f;
  }
  __syncthreads();

  const int row = lane & 15, hi = lane >> 4;
  const v16bf* bp = (const v16bf*)Wpre;
  v8f acc[4];
#pragma unroll
  for (int t = 0; t < 4; ++t) acc[t] = v8f{};

#pragma unroll
  for (int s = 0; s < 2; ++s) {
    v16bf af;
#pragma unroll
    for (int j = 0; j < 16; ++j) af[j] = (__bf16)my[row * 64 + s * 32 + kmap(j, hi)];
#pragma unroll
    for (int t = 0; t < 4; ++t)
      acc[t] = __builtin_amdgcn_wmma_f32_16x16x32_bf16(false, af, false,
                                                       bp[(s * 4 + t) * 32 + lane],
                                                       (short)0, acc[t], false, false);
  }

#pragma unroll
  for (int t = 0; t < 4; ++t) {
    int o = t * 16 + row;
#pragma unroll
    for (int v = 0; v < 8; ++v) {
      int n = nbase + v + hi * 8;
      if (n < N) {
        float val = acc[t][v] + bias[o];
        if (add)     val += add[(size_t)n * 64 + o] * (scale ? scale[n] : 1.0f);
        if (do_silu) val = val / (1.0f + __expf(-val));
        out[(size_t)n * 64 + o] = val;
      }
    }
  }
}

// ------------------------------ pooling / head -----------------------------

__global__ void k_pool_accum(const float* __restrict__ h, const int* __restrict__ batch,
                             float* __restrict__ hg, float* __restrict__ gcnt, int N) {
  int n = blockIdx.x;
  if (n >= N) return;
  int o = threadIdx.x;                                     // 64 threads
  int g = batch[n];
  atomicAdd(&hg[g * 64 + o], h[(size_t)n * 64 + o]);
  if (o == 0) atomicAdd(&gcnt[g], 1.0f);
}

__global__ void k_pool_div(float* __restrict__ hg, const float* __restrict__ gcnt) {
  int t = blockIdx.x * blockDim.x + threadIdx.x;
  if (t < 64 * 64) hg[t] /= fmaxf(gcnt[t >> 6], 1.0f);
}

__global__ void k_head2(const float* __restrict__ z1, const float* __restrict__ w2,
                        const float* __restrict__ b2, float* __restrict__ out) {
  int g = threadIdx.x;
  if (g >= N_GRAPHS) return;
  float s = b2[0];
#pragma unroll
  for (int o = 0; o < 64; ++o) s += z1[g * 64 + o] * w2[o];
  out[g] = s;
}

// ------------------------------- launcher ----------------------------------

extern "C" void kernel_launch(void* const* d_in, const int* in_sizes, int n_in,
                              void* d_out, int out_size, void* d_ws, size_t ws_size,
                              hipStream_t stream) {
  (void)in_sizes; (void)n_in; (void)out_size; (void)ws_size;
  const float* x         = (const float*)d_in[0];
  const float* edge_attr = (const float*)d_in[1];
  const int*   edge_idx  = (const int*)  d_in[2];
  const int*   batch     = (const int*)  d_in[3];
  const float* proj_w    = (const float*)d_in[4];
  const float* proj_b    = (const float*)d_in[5];
  const float* edge_w1   = (const float*)d_in[6];
  const float* edge_b1   = (const float*)d_in[7];
  const float* edge_w2   = (const float*)d_in[8];
  const float* edge_b2   = (const float*)d_in[9];
  const float* root_w    = (const float*)d_in[10];
  const float* conv_b    = (const float*)d_in[11];
  const float* head_w1   = (const float*)d_in[12];
  const float* head_b1   = (const float*)d_in[13];
  const float* head_w2   = (const float*)d_in[14];
  const float* head_b2   = (const float*)d_in[15];
  float* out = (float*)d_out;

  const int N = N_NODES, E = N_EDGES;
  const int* src = edge_idx;
  const int* dst = edge_idx + E;

  // workspace carve-up (256B aligned slabs)
  char* ws = (char*)d_ws;
  size_t off = 0;
  auto alloc = [&](size_t b) -> void* {
    void* p = ws + off;
    off = (off + b + 255) & ~(size_t)255;
    return p;
  };
  float*  deg    = (float*) alloc((size_t)N * 4);
  float*  invdeg = (float*) alloc((size_t)N * 4);
  float*  hA     = (float*) alloc((size_t)N * 64 * 4);
  float*  hB     = (float*) alloc((size_t)N * 64 * 4);
  float*  agg    = (float*) alloc((size_t)N * 64 * 4);
  float*  a1     = (float*) alloc((size_t)E * 64 * 4);
  float*  hg     = (float*) alloc(64 * 64 * 4);
  float*  gcnt   = (float*) alloc(64 * 4);
  float*  z1     = (float*) alloc(64 * 64 * 4);
  const size_t BPRE_ELEMS = (size_t)MSG_STEPS * 4 * 32 * 16;   // per layer
  __bf16* Bpre   = (__bf16*)alloc(LAYERS * BPRE_ELEMS * 2);
  const size_t W64_ELEMS = 2 * 4 * 32 * 16;                    // per 64x64 matrix
  __bf16* WpreP  = (__bf16*)alloc(W64_ELEMS * 2);              // proj
  __bf16* WpreR  = (__bf16*)alloc(LAYERS * W64_ELEMS * 2);     // root x3
  __bf16* WpreH  = (__bf16*)alloc(W64_ELEMS * 2);              // head1

  // --- prep: degree + weight swizzles ---
  k_zero<<<dim3(40), dim3(256), 0, stream>>>(deg, N);
  k_deg<<<dim3((E + 255) / 256), dim3(256), 0, stream>>>(dst, deg, E);
  k_invdeg<<<dim3((N + 255) / 256), dim3(256), 0, stream>>>(deg, invdeg, N);

  k_prep_w64<<<dim3(16), dim3(256), 0, stream>>>(proj_w, WpreP);
  k_prep_w64<<<dim3(16), dim3(256), 0, stream>>>(head_w1, WpreH);
  for (int l = 0; l < LAYERS; ++l) {
    k_prep_w64<<<dim3(16), dim3(256), 0, stream>>>(root_w + (size_t)l * 64 * 64,
                                                   WpreR + (size_t)l * W64_ELEMS);
    k_prep_w2<<<dim3((unsigned)((BPRE_ELEMS + 255) / 256)), dim3(256), 0, stream>>>(
        edge_w2 + (size_t)l * 64 * 4096, edge_b2 + (size_t)l * 4096,
        Bpre + (size_t)l * BPRE_ELEMS);
  }

  // --- input projection: h0 = x @ proj_w + proj_b ---
  const int nodeTiles  = (N + 15) / 16;
  const int nodeBlocks = (nodeTiles + 7) / 8;
  k_node_gemm<<<dim3(nodeBlocks), dim3(256), 0, stream>>>(
      x, WpreP, proj_b, nullptr, nullptr, hA, N, 0);

  // --- layers ---
  float* hcur = hA;
  float* hnxt = hB;
  for (int l = 0; l < LAYERS; ++l) {
    k_edge_mlp1<<<dim3(E), dim3(64), 0, stream>>>(
        edge_attr, edge_w1 + (size_t)l * EDGE_DIM * 64, edge_b1 + (size_t)l * 64, a1, E);
    k_zero<<<dim3(1024), dim3(256), 0, stream>>>(agg, N * 64);
    k_message<<<dim3(E / 16), dim3(32), 0, stream>>>(
        a1, hcur, src, dst, Bpre + (size_t)l * BPRE_ELEMS, agg, E);
    k_node_gemm<<<dim3(nodeBlocks), dim3(256), 0, stream>>>(
        hcur, WpreR + (size_t)l * W64_ELEMS, conv_b + (size_t)l * 64, agg, invdeg,
        hnxt, N, 1);
    float* t = hcur; hcur = hnxt; hnxt = t;
  }

  // --- global mean pool + head ---
  k_zero<<<dim3(17), dim3(256), 0, stream>>>(hg, 64 * 64);
  k_zero<<<dim3(1), dim3(64), 0, stream>>>(gcnt, 64);
  k_pool_accum<<<dim3(N), dim3(64), 0, stream>>>(hcur, batch, hg, gcnt, N);
  k_pool_div<<<dim3(16), dim3(256), 0, stream>>>(hg, gcnt);

  k_node_gemm<<<dim3(1), dim3(256), 0, stream>>>(
      hg, WpreH, head_b1, nullptr, nullptr, z1, 64, 1);
  k_head2<<<dim3(1), dim3(64), 0, stream>>>(z1, head_w2, head_b2, out);
}
